// NormAttention_2388001816998
// MI455X (gfx1250) — matrix operand order repaired and checked by reading, not compiled, not run on hardware
//
#include <hip/hip_runtime.h>
#include <hip/hip_bf16.h>
#include <math.h>

typedef __attribute__((ext_vector_type(16))) _Float16 v16h;
typedef __attribute__((ext_vector_type(8)))  _Float16 v8h;
typedef __attribute__((ext_vector_type(2)))  _Float16 v2h;
typedef __attribute__((ext_vector_type(8)))  float    v8f;

#define WMMA_F16(a, b, c) \
  __builtin_amdgcn_wmma_f32_16x16x32_f16(false, (a), false, (b), (short)0, (c), false, false)

static constexpr int  kB = 2, kN = 2048, kC = 1024, kH = 16, kD = 64;
static constexpr int  kBN = kB * kN;          // 4096 token rows
static constexpr int  k3C = 3 * kC;           // 3072
static constexpr float kScale = 0.125f;       // d^-0.5 = 1/8
static constexpr float kLog2e = 1.4426950408889634f;
static constexpr float kNegBig = -3.0e38f;

__device__ __forceinline__ v16h cat8(v8h lo, v8h hi) {
  v16h r;
#pragma unroll
  for (int i = 0; i < 8; ++i) { r[i] = lo[i]; r[i + 8] = hi[i]; }
  return r;
}

__device__ __forceinline__ float red16_max(float v) {
  v = fmaxf(v, __shfl_xor(v, 1, 32));
  v = fmaxf(v, __shfl_xor(v, 2, 32));
  v = fmaxf(v, __shfl_xor(v, 4, 32));
  v = fmaxf(v, __shfl_xor(v, 8, 32));
  return v;  // xor masks < 16 stay inside each 16-lane half
}
__device__ __forceinline__ float red16_sum(float v) {
  v += __shfl_xor(v, 1, 32);
  v += __shfl_xor(v, 2, 32);
  v += __shfl_xor(v, 4, 32);
  v += __shfl_xor(v, 8, 32);
  return v;
}
__device__ __forceinline__ float red32_sum(float v) {
#pragma unroll
  for (int m = 1; m < 32; m <<= 1) v += __shfl_xor(v, m, 32);
  return v;
}

// ---------------------------------------------------------------------------
// f32 -> f16 conversion (grid-stride)
// ---------------------------------------------------------------------------
__global__ __launch_bounds__(256) void cvt_f16(const float* __restrict__ in,
                                               _Float16* __restrict__ out, int n) {
  for (int i = blockIdx.x * blockDim.x + threadIdx.x; i < n; i += gridDim.x * blockDim.x)
    out[i] = (_Float16)in[i];
}

// ---------------------------------------------------------------------------
// C = A @ Bm^T (+bias).  A: MxK row-major f16.  Bm: NcolsxK row-major f16
// (i.e. Bm holds the already-transposed operand, matching x @ W.T).
// One wave computes a 32x64 strip: 2 A-fragments (row groups) x 4 B-fragments
// shared between them -> 8 WMMAs per 12 b128 loads per K-step of 32.
// ---------------------------------------------------------------------------
__global__ __launch_bounds__(256)
void wmma_gemm_nt(const _Float16* __restrict__ A, const _Float16* __restrict__ Bm,
                  const float* __restrict__ bias, float* __restrict__ C,
                  int M, int Ncols, int K) {
  const int wid  = (blockIdx.x * blockDim.x + threadIdx.x) >> 5;
  const int lane = threadIdx.x & 31;
  const int tiles_n = Ncols >> 6;
  const int tiles_m = M >> 5;                       // 32-row tiles
  if (wid >= tiles_m * tiles_n) return;
  const int mt = wid / tiles_n, nt = wid % tiles_n;
  const int m0 = mt << 5, n0 = nt << 6;
  const int half = lane >> 4, l16 = lane & 15;

  v8f acc00 = {}, acc01 = {}, acc02 = {}, acc03 = {};   // rows m0..m0+15
  v8f acc10 = {}, acc11 = {}, acc12 = {}, acc13 = {};   // rows m0+16..m0+31
  // A fragment: lanes 0-15 -> row +lane, K k0..k0+15;
  //             lanes 16-31 -> same rows,  K k0+16..+31
  const _Float16* arow0 = A + (size_t)(m0 + l16) * K + half * 16;
  const _Float16* arow1 = arow0 + (size_t)16 * K;
  // B fragment: lanes 0-15 -> col n0+lane, K k0..k0+15; lanes 16-31 -> K k0+16..+31
  const _Float16* brow = Bm + (size_t)(n0 + l16) * K + half * 16;
  const size_t bstep = (size_t)16 * K;

  for (int k0 = 0; k0 < K; k0 += 32) {
    // keep the streams warm in WGP$ two K-steps ahead (global_prefetch_b8)
    __builtin_prefetch(brow + k0 + 64, 0, 1);
    __builtin_prefetch(arow0 + k0 + 64, 0, 1);
    v16h a0 = *(const v16h*)(arow0 + k0);
    v16h a1 = *(const v16h*)(arow1 + k0);
    v16h b0 = *(const v16h*)(brow + k0);
    v16h b1 = *(const v16h*)(brow + bstep + k0);
    v16h b2 = *(const v16h*)(brow + 2 * bstep + k0);
    v16h b3 = *(const v16h*)(brow + 3 * bstep + k0);
    acc00 = WMMA_F16(a0, b0, acc00);
    acc10 = WMMA_F16(a1, b0, acc10);
    acc01 = WMMA_F16(a0, b1, acc01);
    acc11 = WMMA_F16(a1, b1, acc11);
    acc02 = WMMA_F16(a0, b2, acc02);
    acc12 = WMMA_F16(a1, b2, acc12);
    acc03 = WMMA_F16(a0, b3, acc03);
    acc13 = WMMA_F16(a1, b3, acc13);
  }
  float b0v = 0.f, b1v = 0.f, b2v = 0.f, b3v = 0.f;
  if (bias) {
    b0v = bias[n0 + l16];      b1v = bias[n0 + 16 + l16];
    b2v = bias[n0 + 32 + l16]; b3v = bias[n0 + 48 + l16];
  }
  float* crow0 = C + (size_t)(m0 + half * 8) * Ncols + n0 + l16;
  float* crow1 = crow0 + (size_t)16 * Ncols;
#pragma unroll
  for (int r = 0; r < 8; ++r) {
    float* cp = crow0 + (size_t)r * Ncols;
    cp[0]  = acc00[r] + b0v;
    cp[16] = acc01[r] + b1v;
    cp[32] = acc02[r] + b2v;
    cp[48] = acc03[r] + b3v;
    float* cq = crow1 + (size_t)r * Ncols;
    cq[0]  = acc10[r] + b0v;
    cq[16] = acc11[r] + b1v;
    cq[32] = acc12[r] + b2v;
    cq[48] = acc13[r] + b3v;
  }
}

// ---------------------------------------------------------------------------
// Per-head LayerNorm of q,k + reorder to [BH,N,64] f16; v -> transposed [BH,64,N]
// One wave handles one (b,h,n) row; 2 elements per lane.
// ---------------------------------------------------------------------------
__global__ __launch_bounds__(256)
void ln_reorder(const float* __restrict__ qkv,
                const float* __restrict__ qg, const float* __restrict__ qb,
                const float* __restrict__ kg, const float* __restrict__ kb,
                _Float16* __restrict__ qh, _Float16* __restrict__ khd,
                _Float16* __restrict__ vT) {
  const int wid  = (blockIdx.x * blockDim.x + threadIdx.x) >> 5;
  const int lane = threadIdx.x & 31;
  if (wid >= kBN * kH) return;
  const int h  = wid & (kH - 1);
  const int bn = wid >> 4;                 // b*N + n
  const int n  = bn & (kN - 1);
  const int bh = (bn >> 11) * kH + h;      // b = bn / 2048
  const int dd = lane << 1;
  const float* src = qkv + (size_t)bn * k3C + h * kD;

  {  // q
    float2 v = *(const float2*)(src + dd);
    float mu = red32_sum(v.x + v.y) * (1.f / 64.f);
    float dx = v.x - mu, dy = v.y - mu;
    float var = red32_sum(dx * dx + dy * dy) * (1.f / 64.f);
    float rstd = rsqrtf(var + 1e-5f);
    v2h o;
    o[0] = (_Float16)(dx * rstd * qg[dd] + qb[dd]);
    o[1] = (_Float16)(dy * rstd * qg[dd + 1] + qb[dd + 1]);
    *(v2h*)(qh + ((size_t)bh * kN + n) * kD + dd) = o;
  }
  {  // k
    float2 v = *(const float2*)(src + kC + dd);
    float mu = red32_sum(v.x + v.y) * (1.f / 64.f);
    float dx = v.x - mu, dy = v.y - mu;
    float var = red32_sum(dx * dx + dy * dy) * (1.f / 64.f);
    float rstd = rsqrtf(var + 1e-5f);
    v2h o;
    o[0] = (_Float16)(dx * rstd * kg[dd] + kb[dd]);
    o[1] = (_Float16)(dy * rstd * kg[dd + 1] + kb[dd + 1]);
    *(v2h*)(khd + ((size_t)bh * kN + n) * kD + dd) = o;
  }
  {  // v transposed
    float2 v = *(const float2*)(src + 2 * kC + dd);
    vT[((size_t)bh * kD + dd) * kN + n]     = (_Float16)v.x;
    vT[((size_t)bh * kD + dd + 1) * kN + n] = (_Float16)v.y;
  }
}

// ---------------------------------------------------------------------------
// Flash-style attention. One wave owns 16 query rows of one (b,h); loops over
// keys in steps of 32 with online softmax. P is re-laid C-layout -> A-layout
// through a per-wave LDS tile (ds stores + s_wait_dscnt + ds b128 loads).
// Output written f16 directly into [B*N, C] order for the final projection.
// ---------------------------------------------------------------------------
__global__ __launch_bounds__(128)
void attn_kernel(const _Float16* __restrict__ qf, const _Float16* __restrict__ kf,
                 const _Float16* __restrict__ vT, const unsigned char* __restrict__ mask,
                 _Float16* __restrict__ Of) {
  __shared__ __align__(16) _Float16 pbuf[4][16 * 32];
  const int wv   = threadIdx.x >> 5;
  const int lane = threadIdx.x & 31;
  const int wid  = blockIdx.x * 4 + wv;
  const int qtiles = kN >> 4;                // 128
  if (wid >= kB * kH * qtiles) return;
  const int bh = wid / qtiles;
  const int q0 = (wid % qtiles) << 4;
  const int b  = bh >> 4, h = bh & 15;
  const int half = lane >> 4, l16 = lane & 15;

  const _Float16* qh = qf + (size_t)bh * kN * kD;
  const _Float16* kh = kf + (size_t)bh * kN * kD;
  const _Float16* vt = vT + (size_t)bh * kD * kN;
  const unsigned char* mrow = mask + (size_t)b * kN * kN;

  // Q fragments (K = d): aq0 covers d 0..31, aq1 covers d 32..63
  v16h aq0, aq1;
  {
    const _Float16* qrow = qh + (size_t)(q0 + l16) * kD + half * 16;
    aq0 = cat8(*(const v8h*)(qrow), *(const v8h*)(qrow + 8));
    aq1 = cat8(*(const v8h*)(qrow + 32), *(const v8h*)(qrow + 40));
  }

  v8f o0 = {}, o1 = {}, o2 = {}, o3 = {};
  float m8[8], l8[8];
#pragma unroll
  for (int r = 0; r < 8; ++r) { m8[r] = kNegBig; l8[r] = 0.f; }
  const float msc = kScale * kLog2e;

  for (int kk0 = 0; kk0 < kN; kk0 += 32) {
    // ---- S = (q @ k^T) for 32 keys: two 16x16 tiles, each K=64 via 2 WMMAs
    v8f s0 = {}, s1 = {};
    {
      const _Float16* kr0 = kh + (size_t)(kk0 + l16) * kD + half * 16;
      const _Float16* kr1 = kh + (size_t)(kk0 + 16 + l16) * kD + half * 16;
      v16h bk;
      bk = *(const v16h*)(kr0);       s0 = WMMA_F16(aq0, bk, s0);
      bk = *(const v16h*)(kr0 + 32);  s0 = WMMA_F16(aq1, bk, s0);
      bk = *(const v16h*)(kr1);       s1 = WMMA_F16(aq0, bk, s1);
      bk = *(const v16h*)(kr1 + 32);  s1 = WMMA_F16(aq1, bk, s1);
    }
    // ---- mask
#pragma unroll
    for (int r = 0; r < 8; ++r) {
      const unsigned char* mr = mrow + (size_t)(q0 + half * 8 + r) * kN + kk0;
      s0[r] = mr[l16]      ? kNegBig : s0[r];
      s1[r] = mr[16 + l16] ? kNegBig : s1[r];
    }
    // ---- online softmax update
    float corr[8];
#pragma unroll
    for (int r = 0; r < 8; ++r) {
      float mx = red16_max(fmaxf(s0[r], s1[r]));
      float mnew = fmaxf(m8[r], mx);
      corr[r] = exp2f((m8[r] - mnew) * msc);
      m8[r] = mnew;
    }
#pragma unroll
    for (int r = 0; r < 8; ++r) {
      o0[r] *= corr[r]; o1[r] *= corr[r]; o2[r] *= corr[r]; o3[r] *= corr[r];
      l8[r] *= corr[r];
    }
    _Float16* pb = pbuf[wv];
#pragma unroll
    for (int r = 0; r < 8; ++r) {
      float p0 = exp2f((s0[r] - m8[r]) * msc);
      float p1 = exp2f((s1[r] - m8[r]) * msc);
      const int lr = half * 8 + r;
      pb[lr * 32 + l16]      = (_Float16)p0;   // keys kk0+0..15
      pb[lr * 32 + 16 + l16] = (_Float16)p1;   // keys kk0+16..31
      l8[r] += red16_sum(p0 + p1);
    }
    asm volatile("s_wait_dscnt 0" ::: "memory");
    // ---- read P back as the A operand (16 q-rows x 32 keys)
    v16h ap;
    {
      const _Float16* pr = pb + l16 * 32 + half * 16;
      ap = cat8(*(const v8h*)(pr), *(const v8h*)(pr + 8));
    }
    // ---- O += P @ V  (B operand from transposed V: K-contiguous per lane)
    {
      const _Float16* vr = vt + (size_t)l16 * kN + kk0 + half * 16;
      const size_t vstep = (size_t)16 * kN;
      v16h bv;
      bv = *(const v16h*)(vr);             o0 = WMMA_F16(ap, bv, o0);
      bv = *(const v16h*)(vr + vstep);     o1 = WMMA_F16(ap, bv, o1);
      bv = *(const v16h*)(vr + 2 * vstep); o2 = WMMA_F16(ap, bv, o2);
      bv = *(const v16h*)(vr + 3 * vstep); o3 = WMMA_F16(ap, bv, o3);
    }
  }
  // ---- epilogue: normalize and scatter into [B*N, C] order
#pragma unroll
  for (int r = 0; r < 8; ++r) {
    float inv = (l8[r] > 0.f) ? 1.f / l8[r] : 0.f;
    _Float16* dst = Of + ((size_t)b * kN + q0 + half * 8 + r) * kC + h * kD + l16;
    dst[0]  = (_Float16)(o0[r] * inv);
    dst[16] = (_Float16)(o1[r] * inv);
    dst[32] = (_Float16)(o2[r] * inv);
    dst[48] = (_Float16)(o3[r] * inv);
  }
}

// ---------------------------------------------------------------------------
extern "C" void kernel_launch(void* const* d_in, const int* in_sizes, int n_in,
                              void* d_out, int out_size, void* d_ws, size_t ws_size,
                              hipStream_t stream) {
  const float*         x      = (const float*)d_in[0];
  const unsigned char* mask   = (const unsigned char*)d_in[1];
  const float*         w_qkv  = (const float*)d_in[2];
  const float*         w_proj = (const float*)d_in[3];
  const float*         b_proj = (const float*)d_in[4];
  const float*         q_g    = (const float*)d_in[5];
  const float*         q_b    = (const float*)d_in[6];
  const float*         k_g    = (const float*)d_in[7];
  const float*         k_b    = (const float*)d_in[8];
  float* out = (float*)d_out;

  // workspace carve-up (256B aligned slabs)
  char* w = (char*)d_ws;
  auto carve = [&](size_t bytes) -> char* {
    char* p = w;
    w += (bytes + 255) & ~(size_t)255;
    return p;
  };
  float*    qkv32  = (float*)   carve((size_t)kBN * k3C * 4);   // 48 MB
  _Float16* x16    = (_Float16*)carve((size_t)kBN * kC * 2);    //  8 MB
  _Float16* wqkv16 = (_Float16*)carve((size_t)k3C * kC * 2);    //  6 MB
  _Float16* wproj16= (_Float16*)carve((size_t)kC * kC * 2);     //  2 MB
  _Float16* q16    = (_Float16*)carve((size_t)kBN * kC * 2);    //  8 MB
  _Float16* k16    = (_Float16*)carve((size_t)kBN * kC * 2);    //  8 MB
  _Float16* vT16   = (_Float16*)carve((size_t)kBN * kC * 2);    //  8 MB
  _Float16* O16    = (_Float16*)carve((size_t)kBN * kC * 2);    //  8 MB

  // 1) f32 -> f16 converts
  cvt_f16<<<1024, 256, 0, stream>>>(x, x16, kBN * kC);
  cvt_f16<<<1024, 256, 0, stream>>>(w_qkv, wqkv16, k3C * kC);
  cvt_f16<<<1024, 256, 0, stream>>>(w_proj, wproj16, kC * kC);

  // 2) qkv = x @ w_qkv^T  (4096 x 3072, K=1024), 32x64 wave tiles
  {
    int waves = (kBN / 32) * (k3C / 64);         // 6144
    wmma_gemm_nt<<<(waves * 32 + 255) / 256, 256, 0, stream>>>(
        x16, wqkv16, nullptr, qkv32, kBN, k3C, kC);
  }

  // 3) LayerNorm(q,k) + reorder + V transpose
  {
    int waves = kBN * kH;                        // 65536
    ln_reorder<<<(waves * 32 + 255) / 256, 256, 0, stream>>>(
        qkv32, q_g, q_b, k_g, k_b, q16, k16, vT16);
  }

  // 4) attention (flash-style, wave per 16 query rows)
  {
    int wtiles = kB * kH * (kN / 16);            // 4096
    attn_kernel<<<wtiles / 4, 128, 0, stream>>>(q16, k16, vT16, mask, O16);
  }

  // 5) out = O @ w_proj^T + b_proj  (4096 x 1024, K=1024), 32x64 wave tiles
  {
    int waves = (kBN / 32) * (kC / 64);          // 2048
    wmma_gemm_nt<<<(waves * 32 + 255) / 256, 256, 0, stream>>>(
        O16, wproj16, b_proj, out, kBN, kC, kC);
  }
}
